// Decoder_20358144983752
// MI455X (gfx1250) — compile-verified, matrix operand order checked by reading, not run on hardware
//
#include <hip/hip_runtime.h>
#include <hip/hip_bf16.h>

typedef __attribute__((ext_vector_type(16))) __bf16 v16bf;
typedef __attribute__((ext_vector_type(8)))  __bf16 v8bf;
typedef __attribute__((ext_vector_type(8)))  float  v8f;

#define D_UNITS 512
#define E_UNITS 512
#define BATCH   64
#define SEQ     2048
#define LDS_ROW 520   // bf16 elements per LDS row (512 + pad), 16B aligned rows
#define M_BLOCK 64    // s-rows per block (4 row-tiles of 16)

// ---------------------------------------------------------------------------
// Pack W1 (f32 [512,512] row-major e->d) into bf16 WMMA B-fragment layout.
// Fragment (ks, ct) covers K = 32*ks..+31, N = 16*ct..+15. Per-lane 16 bf16
// stored contiguously so the GEMM loads each lane's fragment as 32 bytes.
// Lane layout mirrors the ISA 16-bit 32x16 B-matrix table.
// ---------------------------------------------------------------------------
__global__ __launch_bounds__(256) void pack_w1_kernel(
    const float* __restrict__ W1, __bf16* __restrict__ w1p) {
  int idx  = blockIdx.x * 256 + threadIdx.x;  // < 512*512
  int j    = idx & 15;
  int lane = (idx >> 4) & 31;
  int ct   = (idx >> 9) & 31;
  int ks   = idx >> 14;
  int adj  = (lane < 16) ? ((j >= 8) ? 8 : 0) : ((j < 8) ? 8 : 16);
  int k    = ks * 32 + j + adj;          // row of W1 (e index)
  int n    = ct * 16 + (lane & 15);      // col of W1 (d index)
  w1p[idx] = (__bf16)W1[k * D_UNITS + n];
}

// q[b][d] = b1[d] + b2[d] + sum_e hidden[b][e] * W2[e][d]
__global__ __launch_bounds__(256) void q_kernel(
    const float* __restrict__ hidden, const float* __restrict__ W2,
    const float* __restrict__ b1, const float* __restrict__ b2,
    float* __restrict__ q) {
  int idx = blockIdx.x * 256 + threadIdx.x;   // b*512 + d
  int b = idx >> 9, d = idx & 511;
  float acc = b1[d] + b2[d];
  const float* hb = hidden + b * E_UNITS;
  for (int e = 0; e < E_UNITS; ++e)
    acc = fmaf(hb[e], W2[e * D_UNITS + d], acc);
  q[idx] = acc;
}

// ---------------------------------------------------------------------------
// Main fused kernel: scores[b][s] = V . tanh(enc[b,s,:] @ W1 + q[b,:]) + bV
// Block = 8 waves, 64 s-rows x full D=512. bf16 WMMA, f32 accumulate.
// Each wave owns 4 col-tiles across all 4 row-tiles, so every B-fragment
// load feeds 4 WMMAs (quarters L2 traffic on packed W1 vs 1:1).
// feat tensor (256 MB) is never materialized.
// ---------------------------------------------------------------------------
__global__ __launch_bounds__(256) void score_gemm_kernel(
    const float* __restrict__ enc, const __bf16* __restrict__ w1p,
    const float* __restrict__ qbuf, const float* __restrict__ V,
    const float* __restrict__ bV, float* __restrict__ scores) {
  __shared__ __bf16 encLds[M_BLOCK * LDS_ROW];
  __shared__ float  qLds[D_UNITS];
  __shared__ float  scoreBuf[M_BLOCK];

  const int b   = blockIdx.y;
  const int s0  = blockIdx.x * M_BLOCK;
  const int tid = threadIdx.x;

  if (tid < M_BLOCK) scoreBuf[tid] = 0.0f;
  qLds[tid]       = qbuf[b * D_UNITS + tid];
  qLds[tid + 256] = qbuf[b * D_UNITS + 256 + tid];

  // Cooperative tile load: 64 rows x 512 f32 -> bf16 in LDS (8192 float4)
  const float4* enc4 = (const float4*)(enc + ((size_t)(b * SEQ + s0)) * E_UNITS);
  #pragma unroll
  for (int it = 0; it < 32; ++it) {
    int idx  = it * 256 + tid;
    int row  = idx >> 7;
    int col4 = idx & 127;
    float4 v = enc4[idx];
    __bf16* dst = &encLds[row * LDS_ROW + col4 * 4];
    dst[0] = (__bf16)v.x; dst[1] = (__bf16)v.y;
    dst[2] = (__bf16)v.z; dst[3] = (__bf16)v.w;
  }
  __syncthreads();

  const int wc   = tid >> 5;              // wave 0..7 : col group (4 ct = 64 cols)
  const int lane = tid & 31;
  const int half = lane >> 4;             // 0/1
  const int lm   = lane & 15;

  v8f acc[4][4] = {};                     // [row-tile][col-tile]
  #pragma unroll 1
  for (int ks = 0; ks < 16; ++ks) {
    // A fragments for all 4 row-tiles: two 16B LDS loads per lane each
    v16bf af[4];
    #pragma unroll
    for (int rt = 0; rt < 4; ++rt) {
      const __bf16* arow =
          &encLds[(rt * 16 + lm) * LDS_ROW + ks * 32 + half * 8];
      v8bf alo = *(const v8bf*)(arow);        // regs 0-3
      v8bf ahi = *(const v8bf*)(arow + 16);   // regs 4-7
      af[rt] = __builtin_shufflevector(alo, ahi,
          0,1,2,3,4,5,6,7,8,9,10,11,12,13,14,15);
    }
    #pragma unroll
    for (int t = 0; t < 4; ++t) {
      const int ct = wc * 4 + t;
      const v16bf bfrag =
          *(const v16bf*)(w1p + ((ks * 32 + ct) * 32 + lane) * 16);
      #pragma unroll
      for (int rt = 0; rt < 4; ++rt)
        acc[rt][t] = __builtin_amdgcn_wmma_f32_16x16x32_bf16(
            false, af[rt], false, bfrag, (short)0, acc[rt][t], false, false);
    }
  }

  // Fused epilogue: V[n]*tanh(acc+q) summed over this wave's 64 cols
  float part[4][8];
  #pragma unroll
  for (int rt = 0; rt < 4; ++rt)
    #pragma unroll
    for (int v = 0; v < 8; ++v) part[rt][v] = 0.0f;
  #pragma unroll
  for (int t = 0; t < 4; ++t) {
    const int col = (wc * 4 + t) * 16 + lm;
    const float vv = V[col];
    const float qq = qLds[col];
    #pragma unroll
    for (int rt = 0; rt < 4; ++rt)
      #pragma unroll
      for (int v = 0; v < 8; ++v)
        part[rt][v] += vv * tanhf(acc[rt][t][v] + qq);
  }
  // Reduce across the 16 lanes of each half (rows are per-half)
  #pragma unroll
  for (int rt = 0; rt < 4; ++rt) {
    #pragma unroll
    for (int v = 0; v < 8; ++v) {
      float p = part[rt][v];
      p += __shfl_xor(p, 1, 16);
      p += __shfl_xor(p, 2, 16);
      p += __shfl_xor(p, 4, 16);
      p += __shfl_xor(p, 8, 16);
      part[rt][v] = p;
    }
  }
  if (lm == 0) {
    #pragma unroll
    for (int rt = 0; rt < 4; ++rt)
      #pragma unroll
      for (int v = 0; v < 8; ++v)
        atomicAdd(&scoreBuf[rt * 16 + half * 8 + v], part[rt][v]);
  }
  __syncthreads();
  if (tid < M_BLOCK)
    scores[b * SEQ + s0 + tid] = scoreBuf[tid] + bV[0];
}

// Softmax over S per batch; writes attn output directly.
__global__ __launch_bounds__(256) void softmax_kernel(
    const float* __restrict__ scores, float* __restrict__ attn_out) {
  __shared__ float red[256];
  const int b = blockIdx.x, tid = threadIdx.x;
  const float* sc = scores + b * SEQ;
  float v[8], m = -1e30f;
  #pragma unroll
  for (int i = 0; i < 8; ++i) { v[i] = sc[tid + i * 256]; m = fmaxf(m, v[i]); }
  red[tid] = m; __syncthreads();
  for (int off = 128; off > 0; off >>= 1) {
    if (tid < off) red[tid] = fmaxf(red[tid], red[tid + off]);
    __syncthreads();
  }
  m = red[0]; __syncthreads();
  float e[8], s = 0.0f;
  #pragma unroll
  for (int i = 0; i < 8; ++i) { e[i] = __expf(v[i] - m); s += e[i]; }
  red[tid] = s; __syncthreads();
  for (int off = 128; off > 0; off >>= 1) {
    if (tid < off) red[tid] += red[tid + off];
    __syncthreads();
  }
  const float inv = 1.0f / red[0];
  #pragma unroll
  for (int i = 0; i < 8; ++i)
    attn_out[b * SEQ + tid + i * 256] = e[i] * inv;
}

// context[b][e] = sum_s attn[b][s] * enc[b][s][e]  (second pass over enc)
__global__ __launch_bounds__(256) void context_kernel(
    const float* __restrict__ enc, const float* __restrict__ attn,
    float* __restrict__ ctx) {
  __shared__ float wLds[256];
  const int b = blockIdx.y;
  const int e = blockIdx.x * 256 + threadIdx.x;
  float acc = 0.0f;
  for (int c = 0; c < 8; ++c) {
    __syncthreads();
    wLds[threadIdx.x] = attn[b * SEQ + c * 256 + threadIdx.x];
    __syncthreads();
    const float* ep = enc + ((size_t)(b * SEQ + c * 256)) * E_UNITS + e;
    #pragma unroll 4
    for (int i = 0; i < 256; ++i)
      acc = fmaf(wLds[i], ep[(size_t)i * E_UNITS], acc);
  }
  ctx[b * E_UNITS + e] = acc;
}

// xi[b][j] = gru_bias[0][j] + [context | x] @ gru_kernel
__global__ __launch_bounds__(256) void gru_xi_kernel(
    const float* __restrict__ ctx, const float* __restrict__ x,
    const float* __restrict__ gk, const float* __restrict__ gb,
    float* __restrict__ xi) {
  __shared__ float cLds[E_UNITS];
  const int idx = blockIdx.x * 256 + threadIdx.x;  // b*1536 + j
  const int b = idx / 1536, j = idx % 1536;        // 1536 = 6*256: one b/block
  cLds[threadIdx.x]       = ctx[b * E_UNITS + threadIdx.x];
  cLds[threadIdx.x + 256] = ctx[b * E_UNITS + 256 + threadIdx.x];
  __syncthreads();
  float acc = gb[j];
  for (int i = 0; i < E_UNITS; ++i)
    acc = fmaf(cLds[i], gk[i * 1536 + j], acc);
  acc = fmaf(x[b], gk[E_UNITS * 1536 + j], acc);
  xi[idx] = acc;
}

// GRU gates with h0 == 0 (recurrent GEMM collapses to bias), then fc.
__global__ __launch_bounds__(512) void gru_out_kernel(
    const float* __restrict__ xi, const float* __restrict__ gb,
    const float* __restrict__ fcW, const float* __restrict__ fcb,
    float* __restrict__ out_head, float* __restrict__ state_out) {
  __shared__ float red[512];
  const int b = blockIdx.x, j = threadIdx.x;
  const float* xb  = xi + b * 1536;
  const float* gb1 = gb + 1536;
  float z  = 1.0f / (1.0f + __expf(-(xb[j] + gb1[j])));
  float r  = 1.0f / (1.0f + __expf(-(xb[512 + j] + gb1[512 + j])));
  float hh = tanhf(xb[1024 + j] + r * gb1[1024 + j]);
  float st = (1.0f - z) * hh;          // z*h0 term is zero
  state_out[b * D_UNITS + j] = st;
  red[j] = st * fcW[j];
  __syncthreads();
  for (int off = 256; off > 0; off >>= 1) {
    if (j < off) red[j] += red[j + off];
    __syncthreads();
  }
  if (j == 0) out_head[b] = red[0] + fcb[0];
}

extern "C" void kernel_launch(void* const* d_in, const int* in_sizes, int n_in,
                              void* d_out, int out_size, void* d_ws, size_t ws_size,
                              hipStream_t stream) {
  const float* x      = (const float*)d_in[0];
  const float* hidden = (const float*)d_in[1];
  const float* enc    = (const float*)d_in[2];
  const float* W1     = (const float*)d_in[3];
  const float* b1     = (const float*)d_in[4];
  const float* W2     = (const float*)d_in[5];
  const float* b2     = (const float*)d_in[6];
  const float* V      = (const float*)d_in[7];
  const float* bV     = (const float*)d_in[8];
  const float* gk     = (const float*)d_in[9];
  // d_in[10] = gru_rec_kernel: unused, h0 == 0
  const float* gb     = (const float*)d_in[11];
  const float* fcW    = (const float*)d_in[12];
  const float* fcb    = (const float*)d_in[13];

  float* out       = (float*)d_out;
  float* out_head  = out;                           // [64]
  float* out_state = out + BATCH;                   // [64,512]
  float* out_attn  = out + BATCH + BATCH * D_UNITS; // [64,2048]

  char* ws = (char*)d_ws;
  __bf16* w1p   = (__bf16*)(ws);                  // 512 KB
  float*  qbuf  = (float*)(ws + 524288);          // 128 KB
  float*  scores= (float*)(ws + 655360);          // 512 KB
  float*  ctx   = (float*)(ws + 1179648);         // 128 KB
  float*  xi    = (float*)(ws + 1310720);         // 384 KB

  pack_w1_kernel<<<1024, 256, 0, stream>>>(W1, w1p);
  q_kernel<<<128, 256, 0, stream>>>(hidden, W2, b1, b2, qbuf);
  dim3 gB(SEQ / M_BLOCK, BATCH);
  score_gemm_kernel<<<gB, 256, 0, stream>>>(enc, w1p, qbuf, V, bV, scores);
  softmax_kernel<<<BATCH, 256, 0, stream>>>(scores, out_attn);
  dim3 gD(E_UNITS / 256, BATCH);
  context_kernel<<<gD, 256, 0, stream>>>(enc, out_attn, ctx);
  gru_xi_kernel<<<BATCH * 1536 / 256, 256, 0, stream>>>(ctx, x, gk, gb, xi);
  gru_out_kernel<<<BATCH, 512, 0, stream>>>(xi, gb, fcW, fcb, out_head, out_state);
}
